// BaselineMamba_17849884082765
// MI455X (gfx1250) — matrix operand order repaired and checked by reading, not compiled, not run on hardware
//
#include <hip/hip_runtime.h>
#include <hip/hip_bf16.h>
#include <math.h>

// ---------------------------------------------------------------------------
// Shapes (compile-time constants from the reference)
// ---------------------------------------------------------------------------
#define B_  32
#define T_  512
#define DM  512
#define DI  1024
#define DS  16
#define DTR 32
#define DC  4
#define NL  2
#define CELL 256
#define NC  2
#define NTOK (B_ * T_)   // 16384 token rows

typedef __bf16 bf16_t;
typedef __attribute__((ext_vector_type(4)))  __bf16 v4bf;
typedef __attribute__((ext_vector_type(8)))  __bf16 v8bf;
typedef __attribute__((ext_vector_type(16))) __bf16 v16bf;
typedef __attribute__((ext_vector_type(8)))  float  v8f;

union V16U { v16bf v; v8bf h[2]; };

// ---------------------------------------------------------------------------
// WMMA GEMM:  D[Nrows x Mcols] = act( A[Nrows x K](lda) . W[Mcols x K]^T + bias )
// 256 threads = 8 waves arranged 4(row) x 2(col). Block tile = 128 rows x
// (CT*32) cols, K step = 32. Each wave owns a 32-row x (CT*16)-col subtile,
// computed as a 2 x CT grid of 16x16 WMMAs -> 2*CT v_wmma per wave per K-step.
//   CT=4: block 128x128, 8 WMMAs/wave/step (used when M % 128 == 0)
//   CT=2: block 128x64,  4 WMMAs/wave/step (used for M = 64)
// LDS row stride 40 bf16 (80 B) keeps every fragment gather 16B-aligned so
// they lower to ds_load_b128; staging uses global_load_b128 + ds_store_b64
// (f32 -> packed bf16 via v_cvt_pk_bf16_f32).
// act: 0 = none, 1 = relu
// ---------------------------------------------------------------------------
#define LDSTR 40

template<int CT>
__global__ __launch_bounds__(256)
void gemm_bf16_wmma(const float* __restrict__ A, int lda,
                    const float* __restrict__ W,      // Mcols x K row-major
                    const float* __restrict__ bias,   // may be null
                    float* __restrict__ D, int ldd,
                    int Nrows, int Mcols, int K, int act)
{
    constexpr int BCOLS = CT * 32;          // 64 or 128 output cols per block
    __shared__ bf16_t As[128   * LDSTR];    // 10.0 KB
    __shared__ bf16_t Ws[BCOLS * LDSTR];    // 5.0 / 10.0 KB

    const int tid  = threadIdx.x;
    const int lane = tid & 31;
    const int wave = tid >> 5;
    const int half = lane >> 4;     // 0: lanes 0-15, 1: lanes 16-31
    const int lr   = lane & 15;
    const int waveRow = wave >> 1;  // 0..3  (32-row subtile)
    const int waveCol = wave & 1;   // 0..1  (CT*16-col subtile)

    const int rowBase = blockIdx.y * 128;
    const int colBase = blockIdx.x * BCOLS;

    v8f acc[2][CT] = {};

    for (int k0 = 0; k0 < K; k0 += 32) {
        // Stage A tile (128 rows x 32 k): 1024 float4 -> packed bf16x4
        #pragma unroll
        for (int i = 0; i < 4; ++i) {
            int e = tid + i * 256;
            int r = e >> 3, c4 = e & 7;
            float4 f = *(const float4*)&A[(size_t)(rowBase + r) * lda + k0 + c4 * 4];
            v4bf pk = { (bf16_t)f.x, (bf16_t)f.y, (bf16_t)f.z, (bf16_t)f.w };
            *(v4bf*)&As[r * LDSTR + c4 * 4] = pk;
        }
        // Stage W tile (BCOLS out-cols x 32 k): BCOLS*8 float4
        #pragma unroll
        for (int i = 0; i < BCOLS / 32; ++i) {
            int e = tid + i * 256;
            int r = e >> 3, c4 = e & 7;
            float4 f = *(const float4*)&W[(size_t)(colBase + r) * K + k0 + c4 * 4];
            v4bf pk = { (bf16_t)f.x, (bf16_t)f.y, (bf16_t)f.z, (bf16_t)f.w };
            *(v4bf*)&Ws[r * LDSTR + c4 * 4] = pk;
        }
        __syncthreads();

        // Fragment gathers per ISA 7.12.2 (16B ds_load_b128)
        // A 16x32: lane half holds k-groups {half*8..+7} and {16+half*8..+7}
        V16U a[2];
        #pragma unroll
        for (int i = 0; i < 2; ++i) {
            int r = (waveRow * 32 + i * 16 + lr) * LDSTR;
            a[i].h[0] = *(const v8bf*)&As[r + half * 8];
            a[i].h[1] = *(const v8bf*)&As[r + 16 + half * 8];
        }
        // B 32x16: lane half holds k-groups {half*8..+7} and {16+half*8..+7}
        V16U b[CT];
        #pragma unroll
        for (int j = 0; j < CT; ++j) {
            int c = (waveCol * (CT * 16) + j * 16 + lr) * LDSTR;
            b[j].h[0] = *(const v8bf*)&Ws[c + half * 8];
            b[j].h[1] = *(const v8bf*)&Ws[c + 16 + half * 8];
        }

        #pragma unroll
        for (int i = 0; i < 2; ++i)
            #pragma unroll
            for (int j = 0; j < CT; ++j)
                acc[i][j] = __builtin_amdgcn_wmma_f32_16x16x32_bf16(
                                false, a[i].v, false, b[j].v, (short)0,
                                acc[i][j], false, false);

        __syncthreads();
    }

    // C/D layout: VGPR r -> row = r + 8*half, col = lane&15
    #pragma unroll
    for (int j = 0; j < CT; ++j) {
        int col = colBase + waveCol * (CT * 16) + j * 16 + lr;
        float bv = bias ? bias[col] : 0.0f;
        #pragma unroll
        for (int i = 0; i < 2; ++i) {
            int rbase = rowBase + waveRow * 32 + i * 16 + half * 8;
            #pragma unroll
            for (int r = 0; r < 8; ++r) {
                float v = acc[i][j][r] + bv;
                if (act == 1) v = fmaxf(v, 0.0f);
                D[(size_t)(rbase + r) * ldd + col] = v;
            }
        }
    }
}

// ---------------------------------------------------------------------------
// Modality fusion: per-token norms of 3 reps, softmax over norms, weighted sum
// One 256-thread block per token (each thread owns 2 of the 512 features).
// ---------------------------------------------------------------------------
__global__ __launch_bounds__(256)
void fuse_kernel(const float* __restrict__ r0, const float* __restrict__ r1,
                 const float* __restrict__ r2, float* __restrict__ h)
{
    const int t = blockIdx.x;
    const int tid = threadIdx.x;
    __shared__ float red[256];
    __shared__ float s_norm[3];
    __shared__ float s_w[3];

    const float* reps[3] = { r0 + (size_t)t * DM, r1 + (size_t)t * DM, r2 + (size_t)t * DM };
    float v[3][2];
    #pragma unroll
    for (int i = 0; i < 3; ++i) {
        v[i][0] = reps[i][tid];
        v[i][1] = reps[i][tid + 256];
    }
    #pragma unroll
    for (int i = 0; i < 3; ++i) {
        red[tid] = v[i][0] * v[i][0] + v[i][1] * v[i][1];
        __syncthreads();
        for (int s = 128; s > 0; s >>= 1) {
            if (tid < s) red[tid] += red[tid + s];
            __syncthreads();
        }
        if (tid == 0) s_norm[i] = sqrtf(red[0]);
        __syncthreads();
    }
    if (tid == 0) {
        float n0 = s_norm[0], n1 = s_norm[1], n2 = s_norm[2];
        float mx = fmaxf(n0, fmaxf(n1, n2));
        float e0 = expf(n0 - mx), e1 = expf(n1 - mx), e2 = expf(n2 - mx);
        float s = e0 + e1 + e2;
        s_w[0] = e0 / s; s_w[1] = e1 / s; s_w[2] = e2 / s;
    }
    __syncthreads();

    float o0 = 0.f, o1 = 0.f;
    #pragma unroll
    for (int i = 0; i < 3; ++i) {
        float inv = s_w[i] / fmaxf(s_norm[i], 1e-12f);
        o0 += v[i][0] * inv;
        o1 += v[i][1] * inv;
    }
    h[(size_t)t * DM + tid]       = o0;
    h[(size_t)t * DM + tid + 256] = o1;
}

// ---------------------------------------------------------------------------
// xc = silu(xm * conv_w[:, DC-1] + conv_b);  xm is a strided view into xz
// ---------------------------------------------------------------------------
__global__ __launch_bounds__(256)
void silu_conv_kernel(const float* __restrict__ xz,     // token-stride 2*DI, xm at offset 0
                      const float* __restrict__ conv_w, // DI x DC
                      const float* __restrict__ conv_b, // DI
                      float* __restrict__ xc)
{
    size_t i = (size_t)blockIdx.x * blockDim.x + threadIdx.x;
    if (i >= (size_t)NTOK * DI) return;
    size_t n = i >> 10;            // / DI
    int d = (int)(i & (DI - 1));
    float x = xz[n * (2 * DI) + d];
    float tpre = x * conv_w[d * DC + (DC - 1)] + conv_b[d];
    xc[i] = tpre / (1.0f + expf(-tpre));   // silu
}

// ---------------------------------------------------------------------------
// SSM combine for one branch: dt buffer holds (dt_in . dt_w^T + dt_b) already;
// y = xc * ( softplus(dtlin) * bc + Dskip ),   bc = sum(Bs*Cs) per token.
// One block per token; in-place on dt buffer.
// ---------------------------------------------------------------------------
__global__ __launch_bounds__(256)
void ssm_combine_kernel(float* __restrict__ dt,          // N x DI (in: dtlin, out: y)
                        const float* __restrict__ xc,    // N x DI
                        const float* __restrict__ dbl,   // N x (DTR+2*DS) = N x 64
                        const float* __restrict__ Dskip) // DI
{
    const int t = blockIdx.x;
    const int tid = threadIdx.x;
    __shared__ float sBC[2 * DS];
    __shared__ float s_bc;
    if (tid < 2 * DS) sBC[tid] = dbl[(size_t)t * 64 + DTR + tid];
    __syncthreads();
    if (tid == 0) {
        float acc = 0.f;
        #pragma unroll
        for (int s = 0; s < DS; ++s) acc += sBC[s] * sBC[DS + s];
        s_bc = acc;
    }
    __syncthreads();
    const float bc = s_bc;
    #pragma unroll
    for (int j = 0; j < DI / 256; ++j) {
        int d = tid + j * 256;
        size_t idx = (size_t)t * DI + d;
        float dl = dt[idx];
        float sp = (dl > 20.f) ? dl : log1pf(expf(dl));   // softplus
        dt[idx] = xc[idx] * (sp * bc + Dskip[d]);
    }
}

// ---------------------------------------------------------------------------
// y = (yf + yb) * silu(z);  z is a strided view into xz (offset DI)
// ---------------------------------------------------------------------------
__global__ __launch_bounds__(256)
void gate_kernel(float* __restrict__ yf, const float* __restrict__ yb,
                 const float* __restrict__ xz)
{
    size_t i = (size_t)blockIdx.x * blockDim.x + threadIdx.x;
    if (i >= (size_t)NTOK * DI) return;
    size_t n = i >> 10;
    int d = (int)(i & (DI - 1));
    float z = xz[n * (2 * DI) + DI + d];
    float sz = z / (1.0f + expf(-z));
    yf[i] = (yf[i] + yb[i]) * sz;
}

// ---------------------------------------------------------------------------
// Final classifier: logits = tanh(hid . fc2^T + b); out = log_softmax(logits)
// One wave32 per token; shfl_xor reduction.
// ---------------------------------------------------------------------------
__global__ __launch_bounds__(256)
void head_kernel(const float* __restrict__ hid,    // N x CELL
                 const float* __restrict__ fc2_w,  // NC x CELL
                 const float* __restrict__ fc2_b,  // NC
                 float* __restrict__ out)          // N x NC
{
    int wid = blockIdx.x * (blockDim.x >> 5) + (threadIdx.x >> 5);
    if (wid >= NTOK) return;
    int lane = threadIdx.x & 31;
    const float* hrow = hid + (size_t)wid * CELL;
    float a0 = 0.f, a1 = 0.f;
    #pragma unroll
    for (int j = lane; j < CELL; j += 32) {
        float hv = hrow[j];
        a0 += hv * fc2_w[j];
        a1 += hv * fc2_w[CELL + j];
    }
    #pragma unroll
    for (int off = 16; off > 0; off >>= 1) {
        a0 += __shfl_xor(a0, off, 32);
        a1 += __shfl_xor(a1, off, 32);
    }
    if (lane == 0) {
        float l0 = tanhf(a0 + fc2_b[0]);
        float l1 = tanhf(a1 + fc2_b[1]);
        float mx = fmaxf(l0, l1);
        float lse = mx + logf(expf(l0 - mx) + expf(l1 - mx));
        out[(size_t)wid * NC + 0] = l0 - lse;
        out[(size_t)wid * NC + 1] = l1 - lse;
    }
}

// ---------------------------------------------------------------------------
// Host-side orchestration
// ---------------------------------------------------------------------------
static inline void launch_gemm(const float* A, int lda, const float* W,
                               const float* bias, float* D, int ldd,
                               int Nrows, int Mcols, int K, int act,
                               hipStream_t stream)
{
    if (Mcols % 128 == 0) {
        dim3 grid(Mcols / 128, Nrows / 128, 1);
        gemm_bf16_wmma<4><<<grid, 256, 0, stream>>>(A, lda, W, bias, D, ldd,
                                                    Nrows, Mcols, K, act);
    } else {
        dim3 grid(Mcols / 64, Nrows / 128, 1);
        gemm_bf16_wmma<2><<<grid, 256, 0, stream>>>(A, lda, W, bias, D, ldd,
                                                    Nrows, Mcols, K, act);
    }
}

extern "C" void kernel_launch(void* const* d_in, const int* in_sizes, int n_in,
                              void* d_out, int out_size, void* d_ws, size_t ws_size,
                              hipStream_t stream)
{
    (void)in_sizes; (void)n_in; (void)out_size; (void)ws_size;

    const float* text    = (const float*)d_in[0];
    const float* audio   = (const float*)d_in[1];
    /* beats (d_in[2]) is unused by the reference */
    const float* visual  = (const float*)d_in[3];
    const float* W_text  = (const float*)d_in[4];
    const float* b_text  = (const float*)d_in[5];
    const float* W_audio = (const float*)d_in[6];
    const float* b_audio = (const float*)d_in[7];
    const float* W_vis   = (const float*)d_in[8];
    const float* b_vis   = (const float*)d_in[9];
    const float* in_w    = (const float*)d_in[10];
    const float* in_b    = (const float*)d_in[11];
    const float* conv_w  = (const float*)d_in[12];
    const float* conv_b  = (const float*)d_in[13];
    const float* xproj_w = (const float*)d_in[14];
    const float* dt_w    = (const float*)d_in[15];
    const float* dt_b    = (const float*)d_in[16];
    const float* Dskip   = (const float*)d_in[17];
    const float* conv_w_b  = (const float*)d_in[18];
    const float* conv_b_b  = (const float*)d_in[19];
    const float* xproj_w_b = (const float*)d_in[20];
    const float* dt_w_b    = (const float*)d_in[21];
    const float* dt_b_b    = (const float*)d_in[22];
    const float* Dskip_b   = (const float*)d_in[23];
    const float* out_w   = (const float*)d_in[24];
    const float* out_b   = (const float*)d_in[25];
    const float* fc1_w   = (const float*)d_in[26];
    const float* fc1_b   = (const float*)d_in[27];
    const float* fc2_w   = (const float*)d_in[28];
    const float* fc2_b   = (const float*)d_in[29];
    float* out = (float*)d_out;

    // Workspace layout (all float32). U = NTOK*DM elements.
    float* ws = (float*)d_ws;
    const size_t U = (size_t)NTOK * DM;
    float* hA   = ws;                 // N x 512
    float* hB   = ws + U;             // N x 512
    float* repT = ws + 2 * U;         // N x 512 (dead after fuse; reused by xz)
    float* repA = ws + 3 * U;
    float* repV = ws + 4 * U;
    float* xz   = ws + 2 * U;         // N x 2048 (overlaps reps, used after fuse)
    float* xcf  = ws + 6 * U;         // N x 1024
    float* xcb  = ws + 8 * U;         // N x 1024
    float* dtf  = ws + 10 * U;        // N x 1024
    float* dtb  = ws + 12 * U;        // N x 1024
    float* dblf = ws + 14 * U;        // N x 64
    float* dblb = dblf + (size_t)NTOK * 64;
    float* hid  = ws + 14 * U + (size_t)NTOK * 128;  // N x 256

    const size_t EW_TOTAL = (size_t)NTOK * DI;
    const int EW_GRID = (int)((EW_TOTAL + 255) / 256);

    // 1) Input projections (WMMA GEMMs, fused ReLU)
    launch_gemm(text,   768, W_text,  b_text,  repT, DM, NTOK, DM, 768, 1, stream);
    launch_gemm(audio,  512, W_audio, b_audio, repA, DM, NTOK, DM, 512, 1, stream);
    launch_gemm(visual, 256, W_vis,   b_vis,   repV, DM, NTOK, DM, 256, 1, stream);

    // 2) Norm-softmax modality fusion -> hA
    fuse_kernel<<<NTOK, 256, 0, stream>>>(repT, repA, repV, hA);

    float* h_cur = hA;
    float* h_nxt = hB;

    // 3) Mamba layers
    for (int l = 0; l < NL; ++l) {
        const float* inw  = in_w  + (size_t)l * 2 * DI * DM;
        const float* inb  = in_b  + (size_t)l * 2 * DI;
        const float* outw = out_w + (size_t)l * DM * DI;
        const float* outb = out_b + (size_t)l * DM;

        // xz = h . in_w^T + in_b   (N x 2048)
        launch_gemm(h_cur, DM, inw, inb, xz, 2 * DI, NTOK, 2 * DI, DM, 0, stream);

        // xc = silu(xm * cw[:, -1] + cb), both branches
        silu_conv_kernel<<<EW_GRID, 256, 0, stream>>>(
            xz, conv_w   + (size_t)l * DI * DC, conv_b   + (size_t)l * DI, xcf);
        silu_conv_kernel<<<EW_GRID, 256, 0, stream>>>(
            xz, conv_w_b + (size_t)l * DI * DC, conv_b_b + (size_t)l * DI, xcb);

        // dbl = xc . xproj_w^T   (N x 64)  -> CT=2 path
        launch_gemm(xcf, DI, xproj_w   + (size_t)l * 64 * DI, nullptr, dblf, 64,
                    NTOK, 64, DI, 0, stream);
        launch_gemm(xcb, DI, xproj_w_b + (size_t)l * 64 * DI, nullptr, dblb, 64,
                    NTOK, 64, DI, 0, stream);

        // dtlin = dt_in . dt_w^T + dt_b   (A = first 32 cols of dbl, lda=64)
        launch_gemm(dblf, 64, dt_w   + (size_t)l * DI * DTR, dt_b   + (size_t)l * DI,
                    dtf, DI, NTOK, DI, DTR, 0, stream);
        launch_gemm(dblb, 64, dt_w_b + (size_t)l * DI * DTR, dt_b_b + (size_t)l * DI,
                    dtb, DI, NTOK, DI, DTR, 0, stream);

        // y_branch = xc * (softplus(dtlin) * bc + Dskip)   (in-place on dt buf)
        ssm_combine_kernel<<<NTOK, 256, 0, stream>>>(
            dtf, xcf, dblf, Dskip   + (size_t)l * DI);
        ssm_combine_kernel<<<NTOK, 256, 0, stream>>>(
            dtb, xcb, dblb, Dskip_b + (size_t)l * DI);

        // y = (yf + yb) * silu(z)   (in-place on dtf)
        gate_kernel<<<EW_GRID, 256, 0, stream>>>(dtf, dtb, xz);

        // h_next = y . out_w^T + out_b
        launch_gemm(dtf, DI, outw, outb, h_nxt, DM, NTOK, DM, DI, 0, stream);

        float* tmp = h_cur; h_cur = h_nxt; h_nxt = tmp;
    }

    // 4) hid = relu(h . fc1^T + fc1_b)
    launch_gemm(h_cur, DM, fc1_w, fc1_b, hid, CELL, NTOK, CELL, DM, 1, stream);

    // 5) classifier + tanh + log_softmax (wave32 per token)
    head_kernel<<<NTOK / 8, 256, 0, stream>>>(hid, fc2_w, fc2_b, out);
}
